// RNNTJoint_15582141350295
// MI455X (gfx1250) — matrix operand order repaired and checked by reading, not compiled, not run on hardware
//
#include <hip/hip_runtime.h>
#include <hip/hip_bf16.h>
#include <math.h>

typedef __attribute__((ext_vector_type(2))) float v2f;
typedef __attribute__((ext_vector_type(8))) float v8f;
typedef __bf16 v16bf __attribute__((ext_vector_type(16)));
typedef __bf16 v8bf  __attribute__((ext_vector_type(8)));

union bfrag { v16bf v; v8bf h[2]; };

// Problem constants (from reference setup_inputs)
#define B_DIM 8
#define E_DIM 512
#define P_DIM 640
#define J_DIM 640
#define C_DIM 1025
#define T_DIM 256
#define U_DIM 128

// ---- bf16 split helpers (round-to-nearest-even) ---------------------------
__device__ __forceinline__ unsigned short bf16_rne(float x) {
    unsigned u = __float_as_uint(x);
    unsigned r = u + 0x7FFFu + ((u >> 16) & 1u);
    return (unsigned short)(r >> 16);
}
__device__ __forceinline__ float bf16_to_f(unsigned short h) {
    return __uint_as_float((unsigned)h << 16);
}

// ---------------------------------------------------------------------------
// Projection GEMM (exact fp32 via V_WMMA_F32_16X16X4_F32):
//   out[b, m, n] = sum_k in[b, k, m] * W[n, k] + bias[n]
//   in: [B, K, M], W: [N, K], out: [B, M, N]
// grid: (M/16, N/16, B), block: 32 threads (one wave32)
// ---------------------------------------------------------------------------
__global__ __launch_bounds__(32)
void proj_gemm_f32(const float* __restrict__ in, const float* __restrict__ W,
                   const float* __restrict__ bias, float* __restrict__ out,
                   int K, int M, int N) {
    const int l    = threadIdx.x;
    const int half = l >> 4;
    const int lm   = l & 15;
    const int kb   = half * 2;
    const int m0   = blockIdx.x * 16;
    const int n0   = blockIdx.y * 16;
    const int b    = blockIdx.z;

    const float* inB = in + (size_t)b * K * M;
    const float* Wn  = W + (size_t)(n0 + lm) * K;

    v8f acc = {0.f, 0.f, 0.f, 0.f, 0.f, 0.f, 0.f, 0.f};

#pragma unroll 8
    for (int k0 = 0; k0 < K; k0 += 4) {
        v2f a = { inB[(size_t)(k0 + kb) * M + (m0 + lm)],
                  inB[(size_t)(k0 + kb + 1) * M + (m0 + lm)] };
        v2f bb = { Wn[k0 + kb], Wn[k0 + kb + 1] };
        acc = __builtin_amdgcn_wmma_f32_16x16x4_f32(
            false, a, false, bb, (short)0, acc, false, false);
    }

    const float bv = bias[n0 + lm];
    float* op = out + ((size_t)b * M + m0 + 8 * half) * N + (n0 + lm);
#pragma unroll
    for (int r = 0; r < 8; ++r)
        op[(size_t)r * N] = acc[r] + bv;
}

// ---------------------------------------------------------------------------
// One-time split of W_out [C,J] fp32 -> bf16 hi/lo arrays (for bf16x3 GEMM)
// ---------------------------------------------------------------------------
__global__ __launch_bounds__(256)
void split_bf16(const float* __restrict__ src, unsigned short* __restrict__ hi,
                unsigned short* __restrict__ lo, int n4) {
    const int i = blockIdx.x * 256 + threadIdx.x;   // index over float4 groups
    if (i >= n4) return;
    float4 x = ((const float4*)src)[i];
    unsigned short h0 = bf16_rne(x.x), h1 = bf16_rne(x.y),
                   h2 = bf16_rne(x.z), h3 = bf16_rne(x.w);
    unsigned short l0 = bf16_rne(x.x - bf16_to_f(h0)),
                   l1 = bf16_rne(x.y - bf16_to_f(h1)),
                   l2 = bf16_rne(x.z - bf16_to_f(h2)),
                   l3 = bf16_rne(x.w - bf16_to_f(h3));
    uint2 hw = { (unsigned)h0 | ((unsigned)h1 << 16), (unsigned)h2 | ((unsigned)h3 << 16) };
    uint2 lw = { (unsigned)l0 | ((unsigned)l1 << 16), (unsigned)l2 | ((unsigned)l3 << 16) };
    ((uint2*)hi)[i] = hw;
    ((uint2*)lo)[i] = lw;
}

// ---------------------------------------------------------------------------
// Joint kernel (bf16x3 via V_WMMA_F32_16X16X32_BF16):
// one workgroup = one 16-row block (fixed b,t; u0..u0+15), 4 waves.
//   h = relu(e+p) split into bf16 hi/lo in LDS.
//   Each wave register-blocks 4 N-tiles; per K=32 chunk: 4 LDS b128 (A hi/lo,
//   shared over tiles) + 16 global b128 (B hi/lo) + 12 WMMAs.
// 65 N-tiles = 16 groups of 4 (cols 0..1023, round-robin over waves)
// + remainder tile (col 1024) by wave 0. Raw logits -> d_out.
// ---------------------------------------------------------------------------
#define HS2 (J_DIM + 8)   // 648 ushorts/row = 1296 B: 16B-aligned rows, bank-stride 4

__global__ __launch_bounds__(128)
void joint_logits(const float* __restrict__ e,            // [B,T,J]
                  const float* __restrict__ p,            // [B,U,J]
                  const unsigned short* __restrict__ wh,  // [C,J] bf16 hi
                  const unsigned short* __restrict__ wl,  // [C,J] bf16 lo
                  const float* __restrict__ b_out,        // [C]
                  float* __restrict__ out) {              // [B,T,U,C]
    __shared__ __align__(16) unsigned short hs_hi[16 * HS2];
    __shared__ __align__(16) unsigned short hs_lo[16 * HS2];

    const int u0 = blockIdx.x * 16;
    const int t  = blockIdx.y;
    const int b  = blockIdx.z;

    const float* erow = e + ((size_t)b * T_DIM + t) * J_DIM;
    const float* pbas = p + ((size_t)b * U_DIM + u0) * J_DIM;

    // Cooperative h = relu(e + p), bf16 hi/lo split into LDS
    for (int row = 0; row < 16; ++row) {
        const float4* ev4 = (const float4*)erow;
        const float4* pv4 = (const float4*)(pbas + (size_t)row * J_DIM);
        for (int jv = threadIdx.x; jv < J_DIM / 4; jv += 128) {
            float4 ev = ev4[jv], pv = pv4[jv];
            float h0 = fmaxf(ev.x + pv.x, 0.f), h1 = fmaxf(ev.y + pv.y, 0.f);
            float h2 = fmaxf(ev.z + pv.z, 0.f), h3 = fmaxf(ev.w + pv.w, 0.f);
            unsigned short a0 = bf16_rne(h0), a1 = bf16_rne(h1),
                           a2 = bf16_rne(h2), a3 = bf16_rne(h3);
            unsigned short c0 = bf16_rne(h0 - bf16_to_f(a0)),
                           c1 = bf16_rne(h1 - bf16_to_f(a1)),
                           c2 = bf16_rne(h2 - bf16_to_f(a2)),
                           c3 = bf16_rne(h3 - bf16_to_f(a3));
            uint2 hw = { (unsigned)a0 | ((unsigned)a1 << 16), (unsigned)a2 | ((unsigned)a3 << 16) };
            uint2 lw = { (unsigned)c0 | ((unsigned)c1 << 16), (unsigned)c2 | ((unsigned)c3 << 16) };
            *(uint2*)(hs_hi + row * HS2 + 4 * jv) = hw;
            *(uint2*)(hs_lo + row * HS2 + 4 * jv) = lw;
        }
    }
    __syncthreads();

    const int wave = threadIdx.x >> 5;
    const int l    = threadIdx.x & 31;
    const int half = l >> 4;
    const int lm   = l & 15;

    // A fragment per-lane chunk offsets: lanes 0-15 -> K=k0+{0..7}, k0+16+{0..7}
    //                                    lanes16-31 -> +8 on both runs
    const unsigned short* hhl = hs_hi + lm * HS2;
    const unsigned short* hll = hs_lo + lm * HS2;
    const int aoff = 8 * half;
    const int boff = 8 * half;    // B: same K chunks as A (k0+{0..7}, k0+16+{0..7}; lanes16-31 +8)

    float* outb = out + (((size_t)b * T_DIM + t) * U_DIM + u0) * (size_t)C_DIM
                      + (size_t)(8 * half) * C_DIM;

    for (int g = wave; g < 16; g += 4) {
        const int c0 = g * 64;
        const unsigned short* Wh = wh + (size_t)(c0 + lm) * J_DIM;
        const unsigned short* Wl = wl + (size_t)(c0 + lm) * J_DIM;

        v8f acc[4];
#pragma unroll
        for (int q = 0; q < 4; ++q)
            acc[q] = (v8f){0.f, 0.f, 0.f, 0.f, 0.f, 0.f, 0.f, 0.f};

#pragma unroll 2
        for (int k0 = 0; k0 < J_DIM; k0 += 32) {
            bfrag ah, al;
            ah.h[0] = *(const v8bf*)(hhl + k0 + aoff);
            ah.h[1] = *(const v8bf*)(hhl + k0 + 16 + aoff);
            al.h[0] = *(const v8bf*)(hll + k0 + aoff);
            al.h[1] = *(const v8bf*)(hll + k0 + 16 + aoff);
#pragma unroll
            for (int q = 0; q < 4; ++q) {
                const unsigned short* wqh = Wh + (size_t)q * 16 * J_DIM + k0 + boff;
                const unsigned short* wql = Wl + (size_t)q * 16 * J_DIM + k0 + boff;
                bfrag bh, bl;
                bh.h[0] = *(const v8bf*)(wqh); bh.h[1] = *(const v8bf*)(wqh + 16);
                bl.h[0] = *(const v8bf*)(wql); bl.h[1] = *(const v8bf*)(wql + 16);
                acc[q] = __builtin_amdgcn_wmma_f32_16x16x32_bf16(
                    false, ah.v, false, bh.v, (short)0, acc[q], false, false);
                acc[q] = __builtin_amdgcn_wmma_f32_16x16x32_bf16(
                    false, ah.v, false, bl.v, (short)0, acc[q], false, false);
                acc[q] = __builtin_amdgcn_wmma_f32_16x16x32_bf16(
                    false, al.v, false, bh.v, (short)0, acc[q], false, false);
            }
        }

#pragma unroll
        for (int q = 0; q < 4; ++q) {
            const int cq = c0 + q * 16 + lm;                  // <= 1023
            const float bv = b_out[cq];
            float* op = outb + cq;
#pragma unroll
            for (int r = 0; r < 8; ++r)
                op[(size_t)r * C_DIM] = acc[q][r] + bv;       // row = 8*half + r
        }
    }

    // Remainder tile: only column 1024. Wave 0, clamped B rows, masked store.
    if (wave == 0) {
        const unsigned short* Wh = wh + (size_t)(C_DIM - 1) * J_DIM;
        const unsigned short* Wl = wl + (size_t)(C_DIM - 1) * J_DIM;
        v8f acc = {0.f, 0.f, 0.f, 0.f, 0.f, 0.f, 0.f, 0.f};
#pragma unroll 2
        for (int k0 = 0; k0 < J_DIM; k0 += 32) {
            bfrag ah, al, bh, bl;
            ah.h[0] = *(const v8bf*)(hhl + k0 + aoff);
            ah.h[1] = *(const v8bf*)(hhl + k0 + 16 + aoff);
            al.h[0] = *(const v8bf*)(hll + k0 + aoff);
            al.h[1] = *(const v8bf*)(hll + k0 + 16 + aoff);
            bh.h[0] = *(const v8bf*)(Wh + k0 + boff); bh.h[1] = *(const v8bf*)(Wh + k0 + boff + 16);
            bl.h[0] = *(const v8bf*)(Wl + k0 + boff); bl.h[1] = *(const v8bf*)(Wl + k0 + boff + 16);
            acc = __builtin_amdgcn_wmma_f32_16x16x32_bf16(
                false, ah.v, false, bh.v, (short)0, acc, false, false);
            acc = __builtin_amdgcn_wmma_f32_16x16x32_bf16(
                false, ah.v, false, bl.v, (short)0, acc, false, false);
            acc = __builtin_amdgcn_wmma_f32_16x16x32_bf16(
                false, al.v, false, bh.v, (short)0, acc, false, false);
        }
        if (lm == 0) {
            const float bv = b_out[C_DIM - 1];
            float* op = outb + (C_DIM - 1);
#pragma unroll
            for (int r = 0; r < 8; ++r)
                op[(size_t)r * C_DIM] = acc[r] + bv;
        }
    }
}

// ---------------------------------------------------------------------------
// In-place log_softmax over last dim (C=1025). One wave32 per row.
// ---------------------------------------------------------------------------
__global__ __launch_bounds__(256)
void logsoftmax_rows(float* __restrict__ out) {
    constexpr int PER = (C_DIM + 31) / 32;   // 33
    const int wave = threadIdx.x >> 5;
    const int l    = threadIdx.x & 31;
    const long row = (long)blockIdx.x * 8 + wave;

    float* rp = out + (size_t)row * C_DIM;

    float vals[PER];
    float mx = -INFINITY;
#pragma unroll
    for (int i = 0; i < PER; ++i) {
        const int c = l + 32 * i;
        vals[i] = (c < C_DIM) ? rp[c] : -INFINITY;
        mx = fmaxf(mx, vals[i]);
    }
#pragma unroll
    for (int off = 16; off >= 1; off >>= 1)
        mx = fmaxf(mx, __shfl_xor(mx, off, 32));

    float s = 0.f;
#pragma unroll
    for (int i = 0; i < PER; ++i)
        s += __expf(vals[i] - mx);
#pragma unroll
    for (int off = 16; off >= 1; off >>= 1)
        s += __shfl_xor(s, off, 32);

    const float lse = __logf(s) + mx;
#pragma unroll
    for (int i = 0; i < PER; ++i) {
        const int c = l + 32 * i;
        if (c < C_DIM) rp[c] = vals[i] - lse;
    }
}

// ---------------------------------------------------------------------------
extern "C" void kernel_launch(void* const* d_in, const int* in_sizes, int n_in,
                              void* d_out, int out_size, void* d_ws, size_t ws_size,
                              hipStream_t stream) {
    const float* enc    = (const float*)d_in[0];  // [B,E,T]
    const float* dec    = (const float*)d_in[1];  // [B,P,U]
    const float* W_enc  = (const float*)d_in[2];  // [J,E]
    const float* b_enc  = (const float*)d_in[3];  // [J]
    const float* W_pred = (const float*)d_in[4];  // [J,P]
    const float* b_pred = (const float*)d_in[5];  // [J]
    const float* W_out  = (const float*)d_in[6];  // [C,J]
    const float* b_out  = (const float*)d_in[7];  // [C]
    float* out = (float*)d_out;                   // [B,T,U,C]

    // Workspace layout (all 16B-aligned):
    float* e_ws = (float*)d_ws;                                   // [B,T,J] f32
    float* p_ws = e_ws + (size_t)B_DIM * T_DIM * J_DIM;           // [B,U,J] f32
    unsigned short* wh = (unsigned short*)(p_ws + (size_t)B_DIM * U_DIM * J_DIM); // [C,J] bf16
    unsigned short* wl = wh + (size_t)C_DIM * J_DIM;              // [C,J] bf16

    // Split W_out into bf16 hi/lo (one-time per call)
    const int n4 = (C_DIM * J_DIM) / 4;                           // 164000
    split_bf16<<<dim3((n4 + 255) / 256), 256, 0, stream>>>(W_out, wh, wl, n4);

    // e = enc^T @ W_enc^T + b_enc   (M=T, K=E, N=J), exact f32 WMMA
    proj_gemm_f32<<<dim3(T_DIM / 16, J_DIM / 16, B_DIM), 32, 0, stream>>>(
        enc, W_enc, b_enc, e_ws, E_DIM, T_DIM, J_DIM);
    // p = dec^T @ W_pred^T + b_pred (M=U, K=P, N=J)
    proj_gemm_f32<<<dim3(U_DIM / 16, J_DIM / 16, B_DIM), 32, 0, stream>>>(
        dec, W_pred, b_pred, p_ws, P_DIM, U_DIM, J_DIM);

    // logits = relu(e ⊕ p) @ W_out^T + b_out  (bf16x3) -> d_out (raw)
    joint_logits<<<dim3(U_DIM / 16, T_DIM, B_DIM), 128, 0, stream>>>(
        e_ws, p_ws, wh, wl, b_out, out);

    // in-place log_softmax over C
    const long nrows = (long)B_DIM * T_DIM * U_DIM;  // 262144, divisible by 8
    logsoftmax_rows<<<dim3((unsigned)(nrows / 8)), 256, 0, stream>>>(out);
}
